// SIPGModule_36713380446802
// MI455X (gfx1250) — compile-verified, matrix operand order checked by reading, not run on hardware
//
#include <hip/hip_runtime.h>

#define DF 512
#define NN 2048
#define BB 8

typedef __attribute__((ext_vector_type(16))) __bf16        v16bf;
typedef __attribute__((ext_vector_type(8)))  float         v8f;
typedef __attribute__((ext_vector_type(4)))  unsigned int  u32x4;

union FragAB {
    v16bf v;
    u32x4 u[2];
};

__device__ __forceinline__ unsigned short f2bf(float f) {
    unsigned u = __float_as_uint(f);
    unsigned r = 0x7FFFu + ((u >> 16) & 1u);   // round to nearest even
    return (unsigned short)((u + r) >> 16);
}
__device__ __forceinline__ float bf2f(unsigned short h) {
    return __uint_as_float(((unsigned)h) << 16);
}

__device__ __forceinline__ v8f wmma_bf16(const FragAB& a, const FragAB& b, v8f c) {
    return __builtin_amdgcn_wmma_f32_16x16x32_bf16(
        /*neg_a=*/false, a.v, /*neg_b=*/false, b.v,
        /*c_mod=*/(short)0, c, /*reuse_a=*/false, /*reuse_b=*/false);
}

// Load a 16x32 bf16 fragment (A-layout chunking) from a row-major matrix.
// lane row = base_row + (lane&15); K chunks at k+(hi?8:0) and k+16+(hi?8:0).
__device__ __forceinline__ FragAB load_frag_g(const unsigned short* p, int ld,
                                              int row, int k, int hi) {
    FragAB f;
    const unsigned short* r = p + (size_t)row * ld + k + (hi ? 8 : 0);
    f.u[0] = *(const u32x4*)(r);
    f.u[1] = *(const u32x4*)(r + 16);
    return f;
}

// ---------------------------------------------------------------------------
// Kernel 1: convert inputs + weights to bf16
// ---------------------------------------------------------------------------
__global__ void k_convert(const float* __restrict__ It, const float* __restrict__ Is,
                          const float* __restrict__ Wq, const float* __restrict__ Wk,
                          const float* __restrict__ Wv,
                          unsigned short* __restrict__ bIt, unsigned short* __restrict__ bIs,
                          unsigned short* __restrict__ bW,
                          long nX, long nW) {
    long i = (long)blockIdx.x * blockDim.x + threadIdx.x;
    if (i < nX) {
        bIt[i] = f2bf(It[i]);
        bIs[i] = f2bf(Is[i]);
        return;
    }
    long j = i - nX;
    if (j < nW) {
        const long DD = (long)DF * DF;
        int p = (int)(j / DD);
        long rem = j % DD;
        int a = p / 3, role = p % 3;
        const float* src = (role == 0) ? Wq : (role == 1) ? Wk : Wv;
        bW[j] = f2bf(src[(long)a * DD + rem]);
    }
}

// ---------------------------------------------------------------------------
// Kernel 2: projection GEMM  Y = X @ W^T + b  (bf16 in, bf16 out, f32 acc)
// role 2 (V) stored transposed as [DF][NN] for later PV B-fragment loads.
// ---------------------------------------------------------------------------
__global__ void __launch_bounds__(256)
k_proj(const unsigned short* __restrict__ It, const unsigned short* __restrict__ Is,
       const unsigned short* __restrict__ W,
       const float* __restrict__ bq, const float* __restrict__ bk,
       const float* __restrict__ bv,
       unsigned short* __restrict__ qkv) {
    const int z = blockIdx.z;            // b*12 + p
    const int b = z / 12, p = z % 12;
    const int a = p / 3, role = p % 3;
    const bool isTime = (role == 0) ? (a == 0 || a == 2) : (a == 0 || a == 3);

    const unsigned short* X  = (isTime ? It : Is) + (size_t)b * NN * DF;
    const unsigned short* Wp = W + (size_t)p * DF * DF;
    const float* bias = ((role == 0) ? bq : (role == 1) ? bk : bv) + a * DF;
    unsigned short* out = qkv + ((size_t)p * BB + b) * (size_t)NN * DF;

    const int tid  = threadIdx.x;
    const int lane = tid & 31, w = tid >> 5;
    const int t16  = lane & 15, hi = lane >> 4;
    const int wm = w & 3, wn = w >> 2;
    const int m0 = blockIdx.x * 128 + wm * 32;
    const int n0 = blockIdx.y * 64  + wn * 32;

    const v8f zf = {0.f,0.f,0.f,0.f,0.f,0.f,0.f,0.f};
    v8f acc[2][2];
    acc[0][0] = zf; acc[0][1] = zf; acc[1][0] = zf; acc[1][1] = zf;

    for (int k = 0; k < DF; k += 32) {
        FragAB aF[2], bF[2];
        aF[0] = load_frag_g(X,  DF, m0 + t16,      k, hi);
        aF[1] = load_frag_g(X,  DF, m0 + 16 + t16, k, hi);
        bF[0] = load_frag_g(Wp, DF, n0 + t16,      k, hi);
        bF[1] = load_frag_g(Wp, DF, n0 + 16 + t16, k, hi);
        acc[0][0] = wmma_bf16(aF[0], bF[0], acc[0][0]);
        acc[0][1] = wmma_bf16(aF[0], bF[1], acc[0][1]);
        acc[1][0] = wmma_bf16(aF[1], bF[0], acc[1][0]);
        acc[1][1] = wmma_bf16(aF[1], bF[1], acc[1][1]);
    }

    #pragma unroll
    for (int i = 0; i < 2; ++i)
        #pragma unroll
        for (int j = 0; j < 2; ++j) {
            int col = n0 + j * 16 + t16;
            float bb = bias[col];
            #pragma unroll
            for (int v = 0; v < 8; ++v) {
                int row = m0 + i * 16 + v + 8 * hi;      // C layout: M = v (+8 for hi lanes)
                float val = acc[i][j][v] + bb;
                if (role == 2) out[(size_t)col * NN + row] = f2bf(val);   // transposed V
                else           out[(size_t)row * DF + col] = f2bf(val);
            }
        }
}

// ---------------------------------------------------------------------------
// Kernel 3: fused flash attention for one (batch, block, 16-query tile)
// 8 waves: each owns 64 output columns; scores/probs staged in LDS.
// Next-chunk K/V are prefetched (global_prefetch_b8) while this chunk computes.
// ---------------------------------------------------------------------------
__global__ void __launch_bounds__(256)
k_attn(const unsigned short* __restrict__ qkv, unsigned short* __restrict__ AO) {
    const int q0 = blockIdx.x * 16;
    const int a  = blockIdx.y;
    const int b  = blockIdx.z;
    const size_t matsz = (size_t)NN * DF;
    const unsigned short* Q  = qkv + ((size_t)(a * 3 + 0) * BB + b) * matsz;
    const unsigned short* K  = qkv + ((size_t)(a * 3 + 1) * BB + b) * matsz;
    const unsigned short* Vt = qkv + ((size_t)(a * 3 + 2) * BB + b) * matsz; // [DF][NN]
    unsigned short* out = AO + ((size_t)a * BB + b) * matsz;

    __shared__ __attribute__((aligned(16))) float          Sbuf[16 * 256];
    __shared__ __attribute__((aligned(16))) unsigned short Pbuf[16 * 256];
    __shared__ float red[16 * 16];
    __shared__ float rowmax[16], rowsum[16], fac[16];

    const int tid  = threadIdx.x;
    const int lane = tid & 31, w = tid >> 5;
    const int t16  = lane & 15, hi = lane >> 4;

    // Preload all 16 Q fragments (16 rows x full D) once per wave.
    FragAB qf[16];
    {
        const unsigned short* qrow = Q + (size_t)(q0 + t16) * DF + (hi ? 8 : 0);
        #pragma unroll
        for (int ks = 0; ks < 16; ++ks) {
            qf[ks].u[0] = *(const u32x4*)(qrow + ks * 32);
            qf[ks].u[1] = *(const u32x4*)(qrow + ks * 32 + 16);
        }
    }
    if (tid < 16) { rowmax[tid] = -3.0e38f; rowsum[tid] = 0.f; }

    const float scale = 0.044194173824159216f;   // 1/sqrt(512)
    const v8f zf = {0.f,0.f,0.f,0.f,0.f,0.f,0.f,0.f};
    v8f acc[4];
    #pragma unroll
    for (int t = 0; t < 4; ++t) acc[t] = zf;
    __syncthreads();

    for (int ch = 0; ch < NN / 256; ++ch) {
        // --- prefetch next chunk's K rows and Vt segments into cache
        //     (global_prefetch_b8: fire-and-forget, no counters)
        if (ch + 1 < NN / 256) {
            const int nj0 = (ch + 1) * 256;
            const unsigned short* pk = K + (size_t)(nj0 + tid) * DF;   // one K row/thread
            #pragma unroll
            for (int c = 0; c < 8; ++c)
                __builtin_prefetch(pk + c * 64, 0, 3);                 // 8 x 128B
            const unsigned short* pv0 = Vt + (size_t)(tid * 2)     * NN + nj0;
            const unsigned short* pv1 = Vt + (size_t)(tid * 2 + 1) * NN + nj0;
            #pragma unroll
            for (int c = 0; c < 4; ++c) {                              // 2 rows x 512B
                __builtin_prefetch(pv0 + c * 64, 0, 3);
                __builtin_prefetch(pv1 + c * 64, 0, 3);
            }
        }

        // --- S chunk: wave w computes key tiles 2w, 2w+1 of this 256-key chunk
        #pragma unroll
        for (int jj = 0; jj < 2; ++jj) {
            int jt = w * 2 + jj;
            int j0 = ch * 256 + jt * 16;
            const unsigned short* krow = K + (size_t)(j0 + t16) * DF + (hi ? 8 : 0);
            v8f s = zf;
            #pragma unroll
            for (int ks = 0; ks < 16; ++ks) {
                FragAB kf;
                kf.u[0] = *(const u32x4*)(krow + ks * 32);
                kf.u[1] = *(const u32x4*)(krow + ks * 32 + 16);
                s = wmma_bf16(qf[ks], kf, s);
            }
            int col = jt * 16 + t16;
            #pragma unroll
            for (int v = 0; v < 8; ++v)
                Sbuf[(v + 8 * hi) * 256 + col] = s[v] * scale;
        }
        __syncthreads();

        // --- online softmax stats (16 threads per row, 16 cols each)
        const int r = tid & 15, cp = tid >> 4;
        float lm = -3.0e38f;
        #pragma unroll
        for (int c = 0; c < 16; ++c) lm = fmaxf(lm, Sbuf[r * 256 + cp * 16 + c]);
        red[r * 16 + cp] = lm;
        __syncthreads();
        if (tid < 16) {
            float m = rowmax[tid], cm = -3.0e38f;
            #pragma unroll
            for (int c = 0; c < 16; ++c) cm = fmaxf(cm, red[tid * 16 + c]);
            float nm = fmaxf(m, cm);
            fac[tid] = __expf(m - nm);
            rowmax[tid] = nm;
        }
        __syncthreads();
        float ls = 0.f;
        {
            float m = rowmax[r];
            #pragma unroll
            for (int c = 0; c < 16; ++c) {
                int idx = r * 256 + cp * 16 + c;
                float pv = __expf(Sbuf[idx] - m);
                Pbuf[idx] = f2bf(pv);
                ls += pv;
            }
        }
        red[r * 16 + cp] = ls;
        __syncthreads();
        if (tid < 16) {
            float cs = 0.f;
            #pragma unroll
            for (int c = 0; c < 16; ++c) cs += red[tid * 16 + c];
            rowsum[tid] = rowsum[tid] * fac[tid] + cs;
        }
        __syncthreads();

        // --- rescale accumulators, then PV over this chunk
        float fv[8];
        #pragma unroll
        for (int v = 0; v < 8; ++v) fv[v] = fac[v + 8 * hi];
        #pragma unroll
        for (int t = 0; t < 4; ++t)
            #pragma unroll
            for (int v = 0; v < 8; ++v) acc[t][v] *= fv[v];

        #pragma unroll
        for (int ks = 0; ks < 8; ++ks) {
            FragAB pf;
            int c0 = ks * 32 + (hi ? 8 : 0);
            pf.u[0] = *(const u32x4*)&Pbuf[t16 * 256 + c0];
            pf.u[1] = *(const u32x4*)&Pbuf[t16 * 256 + c0 + 16];
            int kb = ch * 256 + ks * 32 + (hi ? 8 : 0);
            #pragma unroll
            for (int t = 0; t < 4; ++t) {
                int o = w * 64 + t * 16 + t16;
                const unsigned short* vrow = Vt + (size_t)o * NN + kb;
                FragAB vf;
                vf.u[0] = *(const u32x4*)(vrow);
                vf.u[1] = *(const u32x4*)(vrow + 16);
                acc[t] = wmma_bf16(pf, vf, acc[t]);
            }
        }
        __syncthreads();
    }

    // --- epilogue: divide by softmax denominator, store bf16
    float rs[8];
    #pragma unroll
    for (int v = 0; v < 8; ++v) rs[v] = rowsum[v + 8 * hi];
    #pragma unroll
    for (int t = 0; t < 4; ++t) {
        int o = w * 64 + t * 16 + t16;
        #pragma unroll
        for (int v = 0; v < 8; ++v) {
            int row = v + 8 * hi;
            out[(size_t)(q0 + row) * DF + o] = f2bf(acc[t][v] / rs[v]);
        }
    }
}

// ---------------------------------------------------------------------------
// Kernel 4: U = FTT + FTS ; V2 = FSS + FST
// ---------------------------------------------------------------------------
__global__ void k_add(const unsigned short* __restrict__ AO,
                      unsigned short* __restrict__ U, unsigned short* __restrict__ V2,
                      long n) {
    long i = (long)blockIdx.x * blockDim.x + threadIdx.x;
    if (i >= n) return;
    U[i]  = f2bf(bf2f(AO[0 * n + i]) + bf2f(AO[2 * n + i]));
    V2[i] = f2bf(bf2f(AO[1 * n + i]) + bf2f(AO[3 * n + i]));
}

// ---------------------------------------------------------------------------
// Kernel 5: out = U @ V2^T + x_origin   (f32 out)
// ---------------------------------------------------------------------------
__global__ void __launch_bounds__(256)
k_final(const unsigned short* __restrict__ U, const unsigned short* __restrict__ V2,
        const float* __restrict__ xo, float* __restrict__ out) {
    const int b = blockIdx.z;
    const unsigned short* Ub = U  + (size_t)b * NN * DF;
    const unsigned short* Vb = V2 + (size_t)b * NN * DF;
    const float* xob = xo + (size_t)b * NN * NN;
    float* ob = out + (size_t)b * NN * NN;

    const int tid  = threadIdx.x;
    const int lane = tid & 31, w = tid >> 5;
    const int t16  = lane & 15, hi = lane >> 4;
    const int wm = w & 3, wn = w >> 2;
    const int m0 = blockIdx.x * 128 + wm * 32;   // rows (n)
    const int n0 = blockIdx.y * 64  + wn * 32;   // cols (m)

    const v8f zf = {0.f,0.f,0.f,0.f,0.f,0.f,0.f,0.f};
    v8f acc[2][2];
    acc[0][0] = zf; acc[0][1] = zf; acc[1][0] = zf; acc[1][1] = zf;

    for (int k = 0; k < DF; k += 32) {
        FragAB aF[2], bF[2];
        aF[0] = load_frag_g(Ub, DF, m0 + t16,      k, hi);
        aF[1] = load_frag_g(Ub, DF, m0 + 16 + t16, k, hi);
        bF[0] = load_frag_g(Vb, DF, n0 + t16,      k, hi);
        bF[1] = load_frag_g(Vb, DF, n0 + 16 + t16, k, hi);
        acc[0][0] = wmma_bf16(aF[0], bF[0], acc[0][0]);
        acc[0][1] = wmma_bf16(aF[0], bF[1], acc[0][1]);
        acc[1][0] = wmma_bf16(aF[1], bF[0], acc[1][0]);
        acc[1][1] = wmma_bf16(aF[1], bF[1], acc[1][1]);
    }

    #pragma unroll
    for (int i = 0; i < 2; ++i)
        #pragma unroll
        for (int j = 0; j < 2; ++j) {
            int col = n0 + j * 16 + t16;
            #pragma unroll
            for (int v = 0; v < 8; ++v) {
                int row = m0 + i * 16 + v + 8 * hi;
                size_t idx = (size_t)row * NN + col;
                ob[idx] = acc[i][j][v] + xob[idx];
            }
        }
}

// ---------------------------------------------------------------------------
extern "C" void kernel_launch(void* const* d_in, const int* in_sizes, int n_in,
                              void* d_out, int out_size, void* d_ws, size_t ws_size,
                              hipStream_t stream) {
    const float* Itime    = (const float*)d_in[0];
    const float* Ispace   = (const float*)d_in[1];
    const float* x_origin = (const float*)d_in[2];
    const float* Wq = (const float*)d_in[3];
    const float* bq = (const float*)d_in[4];
    const float* Wk = (const float*)d_in[5];
    const float* bk = (const float*)d_in[6];
    const float* Wv = (const float*)d_in[7];
    const float* bv = (const float*)d_in[8];
    float* out = (float*)d_out;

    unsigned short* ws = (unsigned short*)d_ws;
    const size_t SZ = (size_t)BB * NN * DF;            // 8M elements per matrix set
    unsigned short* bIt = ws;                          // SZ
    unsigned short* bIs = bIt + SZ;                    // SZ
    unsigned short* bW  = bIs + SZ;                    // 12*DF*DF
    unsigned short* qkv = bW  + (size_t)12 * DF * DF;  // 12*SZ
    unsigned short* AO  = qkv + 12 * SZ;               // 4*SZ
    unsigned short* Ub  = AO  + 4 * SZ;                // SZ
    unsigned short* Vb  = Ub  + SZ;                    // SZ

    const long nX = (long)SZ, nW = 12L * DF * DF;
    const long tot = nX + nW;
    k_convert<<<(int)((tot + 255) / 256), 256, 0, stream>>>(
        Itime, Ispace, Wq, Wk, Wv, bIt, bIs, bW, nX, nW);

    dim3 g2(NN / 128, DF / 64, BB * 12);
    k_proj<<<g2, 256, 0, stream>>>(bIt, bIs, bW, bq, bk, bv, qkv);

    dim3 g3(NN / 16, 4, BB);
    k_attn<<<g3, 256, 0, stream>>>(qkv, AO);

    k_add<<<(int)((SZ + 255) / 256), 256, 0, stream>>>(AO, Ub, Vb, (long)SZ);

    dim3 g5(NN / 128, NN / 64, BB);
    k_final<<<g5, 256, 0, stream>>>(Ub, Vb, x_origin, out);
}
